// HybridOffloadAttention_44229573214694
// MI455X (gfx1250) — matrix operand order repaired and checked by reading, not compiled, not run on hardware
//
#include <hip/hip_runtime.h>

typedef _Float16 half_t;
typedef __attribute__((ext_vector_type(16))) _Float16 v16h;
typedef __attribute__((ext_vector_type(8)))  _Float16 v8h;
typedef __attribute__((ext_vector_type(8)))  float    v8f;

#define HD 128
#define QLEN 4
#define SLEN 4096
#define KEYS_PER_WAVE 512
#define WAVES_PER_BLOCK 4
#define UNITS_PER_BH 16      // 2 sources * 2 S-halves * 4 waves
#define NUM_BH 128           // B*H = 4*32
#define NUM_UNITS (NUM_BH * UNITS_PER_BH)   // 2048

__device__ __forceinline__ v16h make_v16(v8h lo, v8h hi) {
  v16h r;
#pragma unroll
  for (int i = 0; i < 8; ++i) { r[i] = lo[i]; r[i + 8] = hi[i]; }
  return r;
}

// One wave = one (b,h,source,512-key chunk). Flash attention with online
// softmax; Q padded to 16 rows (rows 4..15 are zero -> harmless).
__global__ __launch_bounds__(128) void fa_partial(
    const half_t* __restrict__ qp,
    const half_t* __restrict__ kA, const half_t* __restrict__ vA,
    const half_t* __restrict__ kB, const half_t* __restrict__ vB,
    float* __restrict__ ws)
{
  const int lane  = threadIdx.x & 31;
  const int wave  = threadIdx.x >> 5;
  const int bh    = blockIdx.x >> 2;
  const int blk   = blockIdx.x & 3;
  const int src   = blk >> 1;
  const int shalf = blk & 1;
  const int unit  = bh * UNITS_PER_BH + blk * WAVES_PER_BLOCK + wave;
  const int key0  = shalf * 2048 + wave * KEYS_PER_WAVE;

  const half_t* kptr = (src == 0 ? kA : kB) + (size_t)bh * SLEN * HD;
  const half_t* vptr = (src == 0 ? vA : vB) + (size_t)bh * SLEN * HD;
  const half_t* qrow = qp + (size_t)bh * QLEN * HD;

  const int sel  = lane >> 4;   // which half-wave (A-layout K split)
  const int mrow = lane & 15;

  // B-operand: Q^T, one v16h per 32-wide d-step. Lane l holds column q = l&15
  // (only 0..3 real); halves 0..7 = d {0..7}+8*sel, halves 8..15 = d {16..23}+8*sel.
  v16h bq[4];
#pragma unroll
  for (int ds = 0; ds < 4; ++ds) {
#pragma unroll
    for (int qi = 0; qi < 16; ++qi)
      bq[ds][qi] = (mrow < QLEN) ? qrow[mrow * HD + ds * 32 + sel * 8 + (qi & 7) + (qi >> 3) * 16] : (half_t)0.f;
  }

  v8f o[8];   // O tile [16q x 128d]: o[j][r] = O[q=r+8*sel][d=j*16+(lane&15)]
#pragma unroll
  for (int j = 0; j < 8; ++j)
#pragma unroll
    for (int r = 0; r < 8; ++r) o[j][r] = 0.f;

  float m_run = -1e30f, l_run = 0.f;
  const float scale = 0.088388347648318447f;   // 1/sqrt(128)

  for (int kb = key0; kb < key0 + KEYS_PER_WAVE; kb += 32) {
    // ---- S^T = K * Q^T for 32 keys (two 16-key tiles), accumulate over d ----
    v8f c0, c1;
#pragma unroll
    for (int r = 0; r < 8; ++r) { c0[r] = 0.f; c1[r] = 0.f; }
#pragma unroll
    for (int ds = 0; ds < 4; ++ds) {
      // A-layout 16x32 f16: lane holds row m=lane&15; halves 0..7 = K
      // {0..7}+8*sel, halves 8..15 = K {16..23}+8*sel  (two 16B loads)
      const half_t* r0 = kptr + (size_t)(kb + mrow) * HD + ds * 32 + sel * 8;
      const half_t* r1 = kptr + (size_t)(kb + 16 + mrow) * HD + ds * 32 + sel * 8;
      v16h a0 = make_v16(*(const v8h*)r0, *(const v8h*)(r0 + 16));
      v16h a1 = make_v16(*(const v8h*)r1, *(const v8h*)(r1 + 16));
      c0 = __builtin_amdgcn_wmma_f32_16x16x32_f16(false, a0, false, bq[ds],
                                                  (short)0, c0, false, false);
      c1 = __builtin_amdgcn_wmma_f32_16x16x32_f16(false, a1, false, bq[ds],
                                                  (short)0, c1, false, false);
    }

    // ---- online softmax over this 32-key block ----
    float s0[8], s1[8];
    float vmax = -1e30f;
#pragma unroll
    for (int r = 0; r < 8; ++r) {
      s0[r] = c0[r] * scale; s1[r] = c1[r] * scale;
      vmax = fmaxf(vmax, fmaxf(s0[r], s1[r]));
    }
    vmax = fmaxf(vmax, __shfl_xor(vmax, 16, 32));   // lanes (n, n+16) share q
    float m_new = fmaxf(m_run, vmax);
    float alpha = __expf(m_run - m_new);
    float psum = 0.f;
    v16h ap;  // P in A-layout: C-layout of S^T maps directly (FA2 trick)
#pragma unroll
    for (int r = 0; r < 8; ++r) {
      float p0 = __expf(s0[r] - m_new);
      float p1 = __expf(s1[r] - m_new);
      psum += p0 + p1;
      ap[r] = (half_t)p0; ap[r + 8] = (half_t)p1;
    }
    psum += __shfl_xor(psum, 16, 32);
    l_run = l_run * alpha + psum;
    m_run = m_new;

    // rescale O: its q index is r+8*sel (VGPR row), alpha lives in lane q
    float af[8];
#pragma unroll
    for (int r = 0; r < 8; ++r) af[r] = __shfl(alpha, r + sel * 8, 32);
#pragma unroll
    for (int j = 0; j < 8; ++j)
#pragma unroll
      for (int r = 0; r < 8; ++r) o[j][r] *= af[r];

    // ---- O += P * V : B-operand = V (lane = d column, halves = keys in A-layout K order) ----
#pragma unroll
    for (int j = 0; j < 8; ++j) {
      v16h bv;
#pragma unroll
      for (int s = 0; s < 16; ++s)
        bv[s] = vptr[(size_t)(kb + sel * 8 + (s & 7) + (s >> 3) * 16) * HD + j * 16 + mrow];
      o[j] = __builtin_amdgcn_wmma_f32_16x16x32_f16(false, ap, false, bv,
                                                    (short)0, o[j], false, false);
    }
  }

  // ---- store partial: unnormalized O (q=0..3 only), running m and l ----
  float* wsO = ws;                                  // NUM_UNITS * 512 f32
  float* wsM = ws + (size_t)NUM_UNITS * 512;        // NUM_UNITS * 4
  float* wsL = wsM + (size_t)NUM_UNITS * QLEN;
  if (lane < 16) {
#pragma unroll
    for (int r = 0; r < QLEN; ++r)        // lanes<16: sel=0 -> q = r
#pragma unroll
      for (int j = 0; j < 8; ++j)
        wsO[(size_t)unit * 512 + r * HD + j * 16 + lane] = o[j][r];
  }
  if (lane < QLEN) {                      // lane q holds stats for q
    wsM[unit * QLEN + lane] = m_run;
    wsL[unit * QLEN + lane] = l_run;
  }
}

// Merge 16 partials per (b,h): logsumexp-weighted combine. grid=128, block=128.
__global__ __launch_bounds__(128) void fa_merge(
    const float* __restrict__ ws, half_t* __restrict__ out,
    float* __restrict__ lse_out)
{
  const int bh = blockIdx.x;
  const int d  = threadIdx.x;     // 0..127
  const float* wsO = ws;
  const float* wsM = ws + (size_t)NUM_UNITS * 512;
  const float* wsL = wsM + (size_t)NUM_UNITS * QLEN;

  for (int q = 0; q < QLEN; ++q) {
    float mi[UNITS_PER_BH], lsev[UNITS_PER_BH];
    float mmax = -1e30f;
#pragma unroll
    for (int i = 0; i < UNITS_PER_BH; ++i) {
      float m = wsM[(bh * UNITS_PER_BH + i) * QLEN + q];
      float l = wsL[(bh * UNITS_PER_BH + i) * QLEN + q];
      mi[i] = m;
      lsev[i] = m + __logf(fmaxf(l, 1e-30f));
      mmax = fmaxf(mmax, lsev[i]);
    }
    float ssum = 0.f;
#pragma unroll
    for (int i = 0; i < UNITS_PER_BH; ++i) ssum += __expf(lsev[i] - mmax);
    float lse_new = mmax + __logf(ssum);

    float acc = 0.f;
#pragma unroll
    for (int i = 0; i < UNITS_PER_BH; ++i) {
      // O_unnorm * exp(m_i - lse_new) == (O_unnorm/l_i) * exp(lse_i - lse_new)
      acc += wsO[(size_t)(bh * UNITS_PER_BH + i) * 512 + q * HD + d] *
             __expf(mi[i] - lse_new);
    }
    out[(bh * QLEN + q) * HD + d] = (half_t)acc;
    if (d == 0) lse_out[bh * QLEN + q] = lse_new;
  }
}

extern "C" void kernel_launch(void* const* d_in, const int* in_sizes, int n_in,
                              void* d_out, int out_size, void* d_ws, size_t ws_size,
                              hipStream_t stream) {
  (void)in_sizes; (void)n_in; (void)out_size; (void)ws_size;
  const half_t* q  = (const half_t*)d_in[0];
  const half_t* kA = (const half_t*)d_in[1];
  const half_t* vA = (const half_t*)d_in[2];
  // k_B_gpu/v_B_gpu (d_in[3],[4]) are copy destinations; math uses host values
  const half_t* kB = (const half_t*)d_in[5];
  const half_t* vB = (const half_t*)d_in[6];
  float* ws = (float*)d_ws;

  half_t* out = (half_t*)d_out;
  float*  lse = (float*)((char*)d_out + (size_t)NUM_BH * QLEN * HD * sizeof(half_t));

  fa_partial<<<NUM_BH * 4, WAVES_PER_BLOCK * 32, 0, stream>>>(q, kA, vA, kB, vB, ws);
  fa_merge<<<NUM_BH, HD, 0, stream>>>(ws, out, lse);
}
